// GINActor_31937376813550
// MI455X (gfx1250) — matrix operand order repaired and checked by reading, not compiled, not run on hardware
//
#include <hip/hip_runtime.h>

typedef unsigned short u16;
typedef unsigned int   u32;
typedef unsigned long long u64;

typedef __attribute__((ext_vector_type(16))) __bf16 v16bf;
typedef __attribute__((ext_vector_type(8)))  float  v8f;
typedef __attribute__((ext_vector_type(4)))  u32    v4u;   // trivially-copyable 128-bit chunk

union Frag {           // 32 bytes: one WMMA bf16 A/B operand (16 bf16 per lane)
    v4u   u[2];
    v16bf v;
};

__device__ __forceinline__ u16 f2bf(float f) {
    union { float f; u32 u; } c{f};
    u32 u = c.u;
    return (u16)((u + 0x7FFFu + ((u >> 16) & 1u)) >> 16);   // RNE
}
__device__ __forceinline__ float bf2f(u16 b) {
    union { u32 u; float f; } c{((u32)b) << 16};
    return c.f;
}

// ---- gfx1250 async global->LDS copy (ASYNCcnt-tracked, §15.18.3 op 98) ----
__device__ __forceinline__ void async_b128(const u16* g, u16* l) {
    const u32 lofs = (u32)(size_t)l;   // generic LDS pointer: addr[31:0] = LDS offset
    asm volatile("global_load_async_to_lds_b128 %0, %1, off"
                 :: "v"(lofs), "v"((u64)(size_t)g)
                 : "memory");
}

#ifdef __has_builtin
#if __has_builtin(__builtin_amdgcn_s_wait_asynccnt)
#define WAIT_ASYNC0() __builtin_amdgcn_s_wait_asynccnt(0)
#endif
#endif
#ifndef WAIT_ASYNC0
#define WAIT_ASYNC0() asm volatile("s_wait_asynccnt 0x0" ::: "memory")
#endif

constexpr int BM     = 128;   // rows per block (8 waves x 16)
constexpr int BK     = 32;    // K per step (= WMMA bf16 K)
constexpr int NNODES = 8192;
constexpr int NWRDS  = NNODES / 32;   // bitmask words per adjacency row

// ---- pack adjacency: bits[j*256 + i/32] bit(i%32) = (adj[j][i] != 0) ----
__global__ __launch_bounds__(256) void pack_adj_kernel(const int* __restrict__ adj,
                                                       u32* __restrict__ bits) {
    int w = blockIdx.x * 256 + threadIdx.x;          // one 32-bit word per thread
    const int* p = adj + (size_t)w * 32;             // 32 consecutive ints (row-major)
    u32 m = 0;
#pragma unroll
    for (int b = 0; b < 32; ++b) m |= (p[b] != 0 ? 1u : 0u) << b;
    bits[w] = m;
}

__global__ void zerof_kernel(float* p, int n) {
    int i = blockIdx.x * blockDim.x + threadIdx.x;
    if (i < n) p[i] = 0.0f;
}

__global__ void cvt_f32_bf16_kernel(const float* __restrict__ in, u16* __restrict__ out, int n) {
    int i = blockIdx.x * blockDim.x + threadIdx.x;
    if (i < n) out[i] = f2bf(in[i]);
}

// ---- bf16 transpose: in (R x Cc) -> out (Cc x R); 32x32 LDS tiles ----
__global__ __launch_bounds__(256) void transpose_bf16_kernel(const u16* __restrict__ in,
                                                             u16* __restrict__ out,
                                                             int R, int Cc) {
    __shared__ u16 tile[32][33];
    const int tx = threadIdx.x & 31;
    const int ty = threadIdx.x >> 5;               // 0..7
    const int r0 = blockIdx.x * 32;
    const int c0 = blockIdx.y * 32;
#pragma unroll
    for (int yy = 0; yy < 4; ++yy)
        tile[ty + yy * 8][tx] = in[(size_t)(r0 + ty + yy * 8) * Cc + c0 + tx];
    __syncthreads();
#pragma unroll
    for (int yy = 0; yy < 4; ++yy)
        out[(size_t)(c0 + ty + yy * 8) * R + r0 + tx] = tile[tx][ty + yy * 8];
}

__global__ void bn_finalize_kernel(const float* __restrict__ sum, const float* __restrict__ sumsq,
                                   const float* __restrict__ gamma, const float* __restrict__ beta,
                                   float* __restrict__ s, float* __restrict__ t, int n, float invN) {
    int i = blockIdx.x * blockDim.x + threadIdx.x;
    if (i < n) {
        float mu  = sum[i] * invN;
        float var = sumsq[i] * invN - mu * mu;
        float sc  = gamma[i] * rsqrtf(var + 1e-5f);
        s[i] = sc;
        t[i] = beta[i] - mu * sc;
    }
}

// ---------------------------------------------------------------------------
// Fused WMMA GEMM:  C[M x NC] = post( pre(A)[M x K] @ B[K x NC] + bias )
//   AMODE 0 : A = bf16 row-major (lda); PRE=0 -> async b128 global->LDS copy,
//             PRE=1 -> a = relu(psc[k]*a + psh[k]) fused while staging
//   AMODE 1 : A = (adjacency^T + I) decompressed from bit-packed adj rows:
//             one independent b32 word load per thread, 16-bit expand to bf16
//   BTMODE 0: B supplied PRE-TRANSPOSED bf16 (BT[n][k], row stride K)
//             -> async b128 global->LDS copy
//   BTMODE 1: B = fp32 row-major K x NC weights; transposed+converted in stage
//   STATS 1 : atomic column sum & sumsq of C (f32, pre-bf16)   [BN stats]
//   STATS 2 : atomic column sum of C                           [mean pooling]
// Double-buffered LDS: stage(k+1) overlaps WMMA(k); one barrier per K-step;
// async copies waited per-wave (ASYNCcnt) before the barrier.
// ---------------------------------------------------------------------------
template<int NC, int AMODE, int BTMODE, int PRE, int POSTRELU, int STATS>
__global__ __launch_bounds__(256) void gemm_kernel(
    const u16*  __restrict__ A, const void* __restrict__ B,
    const float* __restrict__ bias,
    const float* __restrict__ psc, const float* __restrict__ psh,
    u16* __restrict__ C, float* __restrict__ st0, float* __restrict__ st1,
    int K, int lda, const u32* __restrict__ adjbits)
{
    __shared__ alignas(16) u16 Atile[2][BM * BK];    // [row][k], k contiguous
    __shared__ alignas(16) u16 Btile[2][NC * BK];    // [n][k],   k contiguous
    __shared__ float scol[NC];
    __shared__ float scol2[NC];

    const int tid   = threadIdx.x;
    const int wave  = tid >> 5;
    const int lane  = tid & 31;
    const int lrow  = lane & 15;
    const int lhalf = lane >> 4;
    const int m0    = blockIdx.x * BM;

    constexpr bool USE_ASYNC = (AMODE == 0 && PRE == 0) || (BTMODE == 0);
    constexpr int NT = NC / 16;
    v8f acc[NT];
    const v8f vzero = {0.f, 0.f, 0.f, 0.f, 0.f, 0.f, 0.f, 0.f};
#pragma unroll
    for (int i = 0; i < NT; ++i) acc[i] = vzero;

    // per-thread constants for the bit-expand staging (AMODE==1):
    // 256 threads -> (kk in 0..31) x (row-group 0..3) x (16-bit half)
    const int xkk  = tid & 31;
    const int xgrp = (tid >> 5) & 3;
    const int xhw  = tid >> 7;               // 0: bits 0..15, 1: bits 16..31
    const int wbase = (m0 >> 5) + xgrp;      // word index within an adjacency row

    auto stage = [&](int buf, int k0s) {
        // -------- A tile --------
        if (AMODE == 1) {
            const int gj   = k0s + xkk;                       // K index == adj row
            const u32 word = adjbits[(size_t)gj * NWRDS + wbase];
            if (k0s + BK < K)                                 // gfx1250 global_prefetch_b8
                __builtin_prefetch(adjbits + (size_t)(gj + BK) * NWRDS + wbase, 0, 1);
#pragma unroll
            for (int b = 0; b < 16; ++b) {
                const int r  = xgrp * 32 + xhw * 16 + b;
                const int gi = m0 + r;
                u32 v = (word >> (xhw * 16 + b)) & 1u;
                if (gi == gj) v += 1u;                        // (A^T + I): fused self term
                Atile[buf][r * BK + xkk] =
                    (u16)(v == 0u ? 0x0000u : (v == 1u ? 0x3F80u : 0x4000u));
            }
        } else if (PRE) {
            const int r  = tid & 127;
            const int hh = tid >> 7;                          // 16 bf16 each
            const u16* Arow = A + (size_t)(m0 + r) * lda + k0s + hh * 16;
#pragma unroll
            for (int e = 0; e < 16; ++e) {
                const int kg = k0s + hh * 16 + e;
                float f = bf2f(Arow[e]);
                f = fmaxf(psc[kg] * f + psh[kg], 0.0f);       // relu(BN(h1)) fused
                Atile[buf][r * BK + hh * 16 + e] = f2bf(f);
            }
        } else {
            // bulk async copy: BM x BK bf16 in 16B chunks (ASYNCcnt-tracked)
            for (int c = tid; c < BM * 4; c += 256) {
                const int r = c >> 2, q = c & 3;
                async_b128(A + (size_t)(m0 + r) * lda + k0s + q * 8,
                           &Atile[buf][r * BK + q * 8]);
            }
        }
        // -------- B tile --------
        if (BTMODE == 0) {
            const u16* BT = (const u16*)B;                    // pre-transposed bf16
            for (int c = tid; c < NC * 4; c += 256) {
                const int n = c >> 2, q = c & 3;
                async_b128(BT + (size_t)n * K + k0s + q * 8,
                           &Btile[buf][n * BK + q * 8]);
            }
        } else {
#pragma unroll 4
            for (int e = tid; e < NC * BK; e += 256) {        // fp32 weights: T + cvt
                const int n = e & (NC - 1);
                const int k = e / NC;
                Btile[buf][n * BK + k] = f2bf(((const float*)B)[(size_t)(k0s + k) * NC + n]);
            }
        }
    };

    stage(0, 0);
    if (USE_ASYNC) WAIT_ASYNC0();
    __syncthreads();

    int cur = 0;
    for (int k0 = 0; k0 < K; k0 += BK) {
        const int nxt = cur ^ 1;
        if (k0 + BK < K) stage(nxt, k0 + BK);     // overlaps with WMMA below

        // ---------------- WMMA: 16x16x32 bf16 ----------------
        // A lanes 0-15: K = 0..7 & 16..23 ; lanes 16-31: K = 8..15 & 24..31 (ISA 7.12.2)
        const int am = wave * 16 + lrow;
        Frag a;
        const v4u* ap = (const v4u*)&Atile[cur][am * BK];
        a.u[0] = ap[lhalf];                       // K = lhalf*8 + 0..7
        a.u[1] = ap[2 + lhalf];                   // K = 16 + lhalf*8 + 0..7
#pragma unroll
        for (int nb = 0; nb < NT; ++nb) {
            Frag b;
            const v4u* bp = (const v4u*)&Btile[cur][(nb * 16 + lrow) * BK];
            b.u[0] = bp[lhalf];                   // K = lhalf*8 + 0..7
            b.u[1] = bp[2 + lhalf];               // K = 16 + lhalf*8 + 0..7
            acc[nb] = __builtin_amdgcn_wmma_f32_16x16x32_bf16(
                false, a.v, false, b.v, (short)0, acc[nb], false, false);
        }

        if (USE_ASYNC) WAIT_ASYNC0();             // own-wave async copies done
        __syncthreads();                          // cur fully read, nxt fully written
        cur = nxt;
    }

    // ---------------- epilogue ----------------
    if (STATS > 0) {
        for (int i = tid; i < NC; i += 256) {
            scol[i] = 0.f;
            if (STATS == 1) scol2[i] = 0.f;
        }
        __syncthreads();
    }

    // C/D layout: VGPR v -> row = v + 8*lhalf, col = lrow (per 16-col tile)
    const int rbase = m0 + wave * 16 + lhalf * 8;
#pragma unroll
    for (int nb = 0; nb < NT; ++nb) {
        const int n  = nb * 16 + lrow;
        const float bv = bias ? bias[n] : 0.0f;
        float s = 0.f, s2 = 0.f;
#pragma unroll
        for (int v = 0; v < 8; ++v) {
            float c = acc[nb][v] + bv;
            if (POSTRELU) c = fmaxf(c, 0.0f);
            C[(size_t)(rbase + v) * NC + n] = f2bf(c);
            if (STATS > 0) { s += c; if (STATS == 1) s2 += c * c; }
        }
        if (STATS > 0) {
            atomicAdd(&scol[n], s);
            if (STATS == 1) atomicAdd(&scol2[n], s2);
        }
    }
    if (STATS > 0) {
        __syncthreads();
        for (int i = tid; i < NC; i += 256) {
            atomicAdd(&st0[i], scol[i]);
            if (STATS == 1) atomicAdd(&st1[i], scol2[i]);
        }
    }
}

// ---- final graph head: pooled(3H) -> relu(Linear 384x384) -> Linear 384x1 -> log_softmax(1) = 0
__global__ __launch_bounds__(256) void final_mlp_kernel(
    const float* __restrict__ pool, const float* __restrict__ wf1, const float* __restrict__ bf1,
    const float* __restrict__ wf2, const float* __restrict__ bf2,
    float* __restrict__ out, float invN)
{
    __shared__ float p[384];
    __shared__ float y1[384];
    __shared__ float red[256];
    const int tid = threadIdx.x;
    for (int i = tid; i < 384; i += 256) p[i] = pool[i] * invN;
    __syncthreads();
    for (int o = tid; o < 384; o += 256) {
        float a = bf1[o];
        for (int k = 0; k < 384; ++k) a += p[k] * wf1[k * 384 + o];
        y1[o] = fmaxf(a, 0.0f);
    }
    __syncthreads();
    float part = 0.f;
    for (int o = tid; o < 384; o += 256) part += y1[o] * wf2[o];
    red[tid] = part;
    __syncthreads();
    for (int s = 128; s > 0; s >>= 1) {
        if (tid < s) red[tid] += red[tid + s];
        __syncthreads();
    }
    if (tid == 0) {
        float h = red[0] + bf2[0];
        out[0] = h - h;   // log_softmax over a single class is exactly 0
    }
}

extern "C" void kernel_launch(void* const* d_in, const int* in_sizes, int n_in,
                              void* d_out, int out_size, void* d_ws, size_t ws_size,
                              hipStream_t stream) {
    (void)in_sizes; (void)n_in; (void)out_size; (void)ws_size;
    const float* features = (const float*)d_in[0];
    const int*   adj      = (const int*)d_in[1];
    const float* w_a[3] = {(const float*)d_in[2],  (const float*)d_in[8],  (const float*)d_in[14]};
    const float* b_a[3] = {(const float*)d_in[3],  (const float*)d_in[9],  (const float*)d_in[15]};
    const float* gm [3] = {(const float*)d_in[4],  (const float*)d_in[10], (const float*)d_in[16]};
    const float* bt [3] = {(const float*)d_in[5],  (const float*)d_in[11], (const float*)d_in[17]};
    const float* w_b[3] = {(const float*)d_in[6],  (const float*)d_in[12], (const float*)d_in[18]};
    const float* b_b[3] = {(const float*)d_in[7],  (const float*)d_in[13], (const float*)d_in[19]};
    const float* wf1 = (const float*)d_in[20];
    const float* bf1 = (const float*)d_in[21];
    const float* wf2 = (const float*)d_in[22];
    const float* bf2 = (const float*)d_in[23];

    // workspace layout (~16.1 MB): bitmask | X | XT | H0 | H1 | stats+pool
    char* ws = (char*)d_ws;
    u32* bits  = (u32*)ws;                                   // 8 MB
    u16* X     = (u16*)(ws + ((size_t)8  << 20));            // 2 MB (N x 128 bf16)
    u16* XT    = (u16*)(ws + ((size_t)10 << 20));            // 2 MB (128 x N bf16)
    u16* H0    = (u16*)(ws + ((size_t)12 << 20));            // 2 MB
    u16* H1    = (u16*)(ws + ((size_t)14 << 20));            // 2 MB
    float* st  = (float*)(ws + ((size_t)16 << 20));
    float* ssum = st, *ssq = st + 128, *bscale = st + 256, *bshift = st + 384;
    float* pool = st + 512;                                  // 3 x 128

    const int N = NNODES;
    const float invN = 1.0f / (float)N;
    dim3 blk(256);

    pack_adj_kernel<<<(N * NWRDS) / 256, blk, 0, stream>>>(adj, bits);
    zerof_kernel<<<2, blk, 0, stream>>>(pool, 384);
    cvt_f32_bf16_kernel<<<(N * 64 + 255) / 256, blk, 0, stream>>>(features, X, N * 64);

    // ----- layer 1 (Hin = 64) -----
    zerof_kernel<<<1, blk, 0, stream>>>(st, 256);
    transpose_bf16_kernel<<<dim3(N / 32, 64 / 32), blk, 0, stream>>>(X, XT, N, 64);
    gemm_kernel<64, 1, 0, 0, 0, 0><<<N / BM, blk, 0, stream>>>(      // H0 = (A^T+I) @ X
        nullptr, XT, nullptr, nullptr, nullptr, H0, nullptr, nullptr, N, 0, bits);
    gemm_kernel<128, 0, 1, 0, 0, 1><<<N / BM, blk, 0, stream>>>(     // H1 = H0 @ W1 + b1, +BN stats
        H0, w_a[0], b_a[0], nullptr, nullptr, H1, ssum, ssq, 64, 64, nullptr);
    bn_finalize_kernel<<<1, 128, 0, stream>>>(ssum, ssq, gm[0], bt[0], bscale, bshift, 128, invN);
    gemm_kernel<128, 0, 1, 1, 1, 2><<<N / BM, blk, 0, stream>>>(     // X = relu(relu(BN(H1)) @ W2 + b2), +pool
        H1, w_b[0], b_b[0], bscale, bshift, X, pool, nullptr, 128, 128, nullptr);

    // ----- layers 2..3 (Hin = 128) -----
    for (int L = 1; L < 3; ++L) {
        zerof_kernel<<<1, blk, 0, stream>>>(st, 256);
        transpose_bf16_kernel<<<dim3(N / 32, 128 / 32), blk, 0, stream>>>(X, XT, N, 128);
        gemm_kernel<128, 1, 0, 0, 0, 0><<<N / BM, blk, 0, stream>>>(
            nullptr, XT, nullptr, nullptr, nullptr, H0, nullptr, nullptr, N, 0, bits);
        gemm_kernel<128, 0, 1, 0, 0, 1><<<N / BM, blk, 0, stream>>>(
            H0, w_a[L], b_a[L], nullptr, nullptr, H1, ssum, ssq, 128, 128, nullptr);
        bn_finalize_kernel<<<1, 128, 0, stream>>>(ssum, ssq, gm[L], bt[L], bscale, bshift, 128, invN);
        gemm_kernel<128, 0, 1, 1, 1, 2><<<N / BM, blk, 0, stream>>>(
            H1, w_b[L], b_b[L], bscale, bshift, X, pool + 128 * L, nullptr, 128, 128, nullptr);
    }

    final_mlp_kernel<<<1, blk, 0, stream>>>(pool, wf1, bf1, wf2, bf2, (float*)d_out, invN);
}
